// KANLayer_70334384439341
// MI455X (gfx1250) — matrix operand order repaired and checked by reading, not compiled, not run on hardware
//
#include <hip/hip_runtime.h>

// KAN layer (PCHIP spline mixture) for MI455X / gfx1250.
// Strategy: per input-dim i, the spline evaluation is a sparse-A dense-B GEMM:
//   y[b,o] += A_c[b,:] * ct[i,:,o] + A_d[b,:] * dt[i,:,o]   (K = 64 knots)
// A has 2 nonzeros per row (Hermite weights at idx, idx+1); extrapolation is
// weight 1 / (x - X) at the end knots. We run it densely on the WMMA units:
// v_wmma_f32_16x16x32_f16, f32 accumulation, f16 tables staged in [i][o][k]
// order (the exact B-fragment layout). Tables are 4 MB f16 -> resident in L2.

#define KK      64
#define DIN     128
#define DOUT    128
#define NROWS   4096
#define EPSF    1e-12f

typedef __attribute__((ext_vector_type(16))) _Float16 v16h;
typedef __attribute__((ext_vector_type(8)))  _Float16 v8h;
typedef __attribute__((ext_vector_type(8)))  float    v8f;

// Concatenate two 16-byte half-runs into one WMMA 16-half fragment.
__device__ __forceinline__ v16h cat8(const _Float16* p0, const _Float16* p1) {
    v8h lo = *(const v8h*)p0;
    v8h hi = *(const v8h*)p1;
    v16h r;
#pragma unroll
    for (int e = 0; e < 8; ++e) { r[e] = lo[e]; r[e + 8] = hi[e]; }
    return r;
}

// ---------------------------------------------------------------------------
// Kernel 1: PCHIP slopes from coeffs, then emit f16 tables transposed to
// Tc[i][o][k] = coeffs[o][i][k], Td[i][o][k] = slopes[o][i][k].
// One thread per (o,i) pair: 16384 threads, trivial cost.
// ---------------------------------------------------------------------------
__global__ __launch_bounds__(256) void kan_prep(const float* __restrict__ coeffs,
                                                const float* __restrict__ knots,
                                                _Float16* __restrict__ Tc,
                                                _Float16* __restrict__ Td) {
    int tid = blockIdx.x * 256 + threadIdx.x;
    if (tid >= DOUT * DIN) return;
    int o  = tid >> 7;
    int ii = tid & 127;
    const float* y = coeffs + ((size_t)(o * DIN + ii)) * KK;

    float h[KK - 1], delta[KK - 1], d[KK];
#pragma unroll 1
    for (int k = 0; k < KK - 1; ++k) {
        h[k]     = knots[k + 1] - knots[k];
        delta[k] = (y[k + 1] - y[k]) / (h[k] + EPSF);
    }
#pragma unroll 1
    for (int k = 1; k < KK - 1; ++k) {
        float h0 = h[k - 1], h1 = h[k];
        float w1 = 2.f * h1 + h0, w2 = h1 + 2.f * h0;
        float d0 = delta[k - 1], d1 = delta[k];
        float denom = w1 / (d0 + EPSF) + w2 / (d1 + EPSF);
        d[k] = (d0 * d1 > 0.f) ? (w1 + w2) / (denom + EPSF) : 0.f;
    }
    {   // first endpoint + limiter
        float df = ((2.f * h[0] + h[1]) * delta[0] - h[0] * delta[1]) /
                   (h[0] + h[1] + EPSF);
        if (df * delta[0] <= 0.f) df = 0.f;
        if (fabsf(df) > 3.f * fabsf(delta[0])) df = 3.f * delta[0];
        d[0] = df;
    }
    {   // last endpoint + limiter
        const int L = KK - 2;
        float dl = ((2.f * h[L] + h[L - 1]) * delta[L] - h[L] * delta[L - 1]) /
                   (h[L] + h[L - 1] + EPSF);
        if (dl * delta[L] <= 0.f) dl = 0.f;
        if (fabsf(dl) > 3.f * fabsf(delta[L])) dl = 3.f * delta[L];
        d[KK - 1] = dl;
    }
    _Float16* tc = Tc + ((size_t)(ii * DOUT + o)) * KK;
    _Float16* td = Td + ((size_t)(ii * DOUT + o)) * KK;
#pragma unroll 1
    for (int k = 0; k < KK; ++k) {
        tc[k] = (_Float16)y[k];
        td[k] = (_Float16)d[k];
    }
}

// ---------------------------------------------------------------------------
// Kernel 2: main WMMA evaluation.
// Block = 16 batch rows, 8 waves; wave w owns output columns [16w, 16w+16).
// Per i: build sparse A tiles (coeff + slope weights) in LDS (double-buffered),
// then 4 chained v_wmma_f32_16x16x32_f16 per wave into a v8f accumulator.
// ---------------------------------------------------------------------------
__global__ __launch_bounds__(256) void kan_main(const float* __restrict__ x,
                                                const float* __restrict__ bias,
                                                const _Float16* __restrict__ Tc,
                                                const _Float16* __restrict__ Td,
                                                float* __restrict__ out) {
    // A buffers: [ping-pong][table: 0=coeff,1=slope][row][k]  -> 8 KB
    __shared__ _Float16 Ab[2][2][16][KK];
    __shared__ float    xs[16 * DIN];          // 8 KB batch-tile of x

    const int tidx = threadIdx.x;
    const int lane = tidx & 31;
    const int wave = tidx >> 5;                // 0..7 -> N tile
    const int hs   = lane >> 4;                // lane half-group
    const int lm   = lane & 15;
    const int b0   = blockIdx.x * 16;
    const int n0   = wave * 16;

    // Stage x tile (coalesced: contiguous 2048 floats)
    for (int j = tidx; j < 16 * DIN; j += 256) xs[j] = x[(size_t)b0 * DIN + j];

    const float H    = 1.0f / (KK - 1);        // knot spacing
    const float RINV = (float)(KK - 1);        // 1/H, exact constant: no division

    // Fold bias in: all 8 acc elements are the same output column n0+lm.
    const float bv = bias[n0 + lm];
    v8f acc;
#pragma unroll
    for (int v = 0; v < 8; ++v) acc[v] = bv;

    for (int i = 0; i < DIN; ++i) {
        const int buf = i & 1;

        // Zero this A buffer: 2*16*64 halves = 1024 dwords / 256 threads.
        unsigned int* z = (unsigned int*)&Ab[buf][0][0][0];
#pragma unroll
        for (int t = 0; t < 4; ++t) z[tidx + t * 256] = 0u;
        __syncthreads();

        // Scatter Hermite weights (threads 0..15, one per batch row).
        if (tidx < 16) {
            float xv = xs[tidx * DIN + i];
            _Float16* ac = &Ab[buf][0][tidx][0];
            _Float16* ad = &Ab[buf][1][tidx][0];
            if (xv < 0.0f) {                       // left linear extrapolation
                ac[0] = (_Float16)1.0f;  ad[0] = (_Float16)xv;
            } else if (xv > 1.0f) {                // right linear extrapolation
                ac[KK - 1] = (_Float16)1.0f;  ad[KK - 1] = (_Float16)(xv - 1.0f);
            } else {
                // s = x/H; idx = floor(s); t = frac -- multiply, no division.
                float s  = xv * RINV;
                int idx  = (int)floorf(s);
                idx = idx < 0 ? 0 : (idx > KK - 2 ? KK - 2 : idx);
                float t  = s - (float)idx;
                float t2 = t * t, t3 = t2 * t;
                float h00 =  2.f * t3 - 3.f * t2 + 1.f;
                float h10 =        t3 - 2.f * t2 + t;
                float h01 = -2.f * t3 + 3.f * t2;
                float h11 =        t3 -       t2;
                ac[idx]     = (_Float16)h00;
                ac[idx + 1] = (_Float16)h01;
                ad[idx]     = (_Float16)(h10 * H);
                ad[idx + 1] = (_Float16)(h11 * H);
            }
        }
        __syncthreads();

        // A fragments from LDS (contiguous k order, same as B: lane half-groups read
        // a contiguous 16-half K-run at 32*chunk + 16*hs).
        const _Float16* arc = &Ab[buf][0][lm][0];
        const _Float16* ard = &Ab[buf][1][lm][0];
        v16h ac0 = cat8(arc +      16 * hs, arc +      16 * hs + 8);
        v16h ac1 = cat8(arc + 32 + 16 * hs, arc + 32 + 16 * hs + 8);
        v16h ad0 = cat8(ard +      16 * hs, ard +      16 * hs + 8);
        v16h ad1 = cat8(ard + 32 + 16 * hs, ard + 32 + 16 * hs + 8);

        // B fragments from global tables [i][o][k]: lane = column n0+lm,
        // contiguous 16-half K-run at 32*chunk + 16*hs.
        const _Float16* bc = Tc + ((size_t)i * DOUT + n0 + lm) * KK;
        const _Float16* bd = Td + ((size_t)i * DOUT + n0 + lm) * KK;
        v16h bc0 = cat8(bc +      16 * hs, bc +      16 * hs + 8);
        v16h bc1 = cat8(bc + 32 + 16 * hs, bc + 32 + 16 * hs + 8);
        v16h bd0 = cat8(bd +      16 * hs, bd +      16 * hs + 8);
        v16h bd1 = cat8(bd + 32 + 16 * hs, bd + 32 + 16 * hs + 8);

        // Prefetch next i's table rows (global_prefetch_b8).
        if (i + 1 < DIN) {
            __builtin_prefetch(bc + (size_t)DOUT * KK, 0, 1);
            __builtin_prefetch(bd + (size_t)DOUT * KK, 0, 1);
        }

        // D = A*B + C, chained f32 accumulation.
        acc = __builtin_amdgcn_wmma_f32_16x16x32_f16(false, ac0, false, bc0,
                                                     (short)0, acc, false, false);
        acc = __builtin_amdgcn_wmma_f32_16x16x32_f16(false, ac1, false, bc1,
                                                     (short)0, acc, false, false);
        acc = __builtin_amdgcn_wmma_f32_16x16x32_f16(false, ad0, false, bd0,
                                                     (short)0, acc, false, false);
        acc = __builtin_amdgcn_wmma_f32_16x16x32_f16(false, ad1, false, bd1,
                                                     (short)0, acc, false, false);
    }

    // Epilogue: C/D layout is M = v + 8*hs, N = lm. Bias already folded in.
#pragma unroll
    for (int v = 0; v < 8; ++v) {
        int row = b0 + v + 8 * hs;
        out[(size_t)row * DOUT + n0 + lm] = acc[v];
    }
}

// ---------------------------------------------------------------------------
// Launch: inputs are (x, coeffs, bias, knots), all float32.
// Workspace: Tc (2 MB f16) + Td (2 MB f16) = 4 MB.
// ---------------------------------------------------------------------------
extern "C" void kernel_launch(void* const* d_in, const int* in_sizes, int n_in,
                              void* d_out, int out_size, void* d_ws, size_t ws_size,
                              hipStream_t stream) {
    const float* x      = (const float*)d_in[0];
    const float* coeffs = (const float*)d_in[1];
    const float* bias   = (const float*)d_in[2];
    const float* knots  = (const float*)d_in[3];

    _Float16* Tc = (_Float16*)d_ws;
    _Float16* Td = Tc + (size_t)DIN * DOUT * KK;

    kan_prep<<<(DOUT * DIN + 255) / 256, 256, 0, stream>>>(coeffs, knots, Tc, Td);
    kan_main<<<NROWS / 16, 256, 0, stream>>>(x, bias, Tc, Td, (float*)d_out);
}
